// Decoder_52630529245342
// MI455X (gfx1250) — matrix operand order repaired and checked by reading, not compiled, not run on hardware
//
#include <hip/hip_runtime.h>
#include <hip/hip_bf16.h>
#include <math.h>

// ---------------- problem constants ----------------
#define BB      32
#define TT      512
#define EPROJS  512
#define DUNITS  1024
#define ODIM    10000
#define ATT     320
#define LL      128
#define LP1     129            // L+1 decode steps
#define NROWS   (LP1 * BB)     // 4128 rows for output projection
#define SOS_EOS (ODIM - 1)
#define K0CAT   (DUNITS + EPROJS + DUNITS)  // 2560  = [ey | att_c | z0]
#define K1CAT   (DUNITS + DUNITS)           // 2048  = [z0new | z1]
#define G4      (4 * DUNITS)                // 4096 gates
#define GPLANE  (BB * G4)                   // one K-split partial plane

typedef __attribute__((ext_vector_type(16))) __bf16 v16bf;
typedef __attribute__((ext_vector_type(8)))  __bf16 v8bf;
typedef __attribute__((ext_vector_type(8)))  float  v8f;

__device__ __forceinline__ float sigmoidf_(float x) { return 1.0f / (1.0f + __expf(-x)); }

// build one 16-lane-resident WMMA bf16 fragment from two contiguous 8-elem chunks
__device__ __forceinline__ v16bf load_frag(const __bf16* p0, const __bf16* p1) {
    v8bf lo = *(const v8bf*)p0;
    v8bf hi = *(const v8bf*)p1;
    v16bf r;
#pragma unroll
    for (int i = 0; i < 8; ++i) { r[i] = lo[i]; r[i + 8] = hi[i]; }
    return r;
}

// ============================================================
// init: zero LSTM state + loss accumulators (every launch)
// ============================================================
__global__ void init_state_k(float* z0, float* c0, float* z1, float* c1, float* accum) {
    int i = blockIdx.x * blockDim.x + threadIdx.x;
    if (i < BB * DUNITS) { z0[i] = 0.f; c0[i] = 0.f; z1[i] = 0.f; c1[i] = 0.f; }
    if (i < 2) accum[i] = 0.f;
}

// ============================================================
// weight packing: f32 -> bf16, [W_ih | W_hh] concatenated along K
// ============================================================
__global__ void build_w0cat_k(const float* __restrict__ Wih, const float* __restrict__ Whh,
                              __bf16* __restrict__ Wc) {
    size_t i = (size_t)blockIdx.x * blockDim.x + threadIdx.x;
    if (i >= (size_t)G4 * K0CAT) return;
    int n = (int)(i / K0CAT), k = (int)(i % K0CAT);
    float v = (k < DUNITS + EPROJS) ? Wih[(size_t)n * (DUNITS + EPROJS) + k]
                                    : Whh[(size_t)n * DUNITS + (k - (DUNITS + EPROJS))];
    Wc[i] = (__bf16)v;
}

__global__ void build_w1cat_k(const float* __restrict__ Wih, const float* __restrict__ Whh,
                              __bf16* __restrict__ Wc) {
    size_t i = (size_t)blockIdx.x * blockDim.x + threadIdx.x;
    if (i >= (size_t)G4 * K1CAT) return;
    int n = (int)(i / K1CAT), k = (int)(i % K1CAT);
    float v = (k < DUNITS) ? Wih[(size_t)n * DUNITS + k]
                           : Whh[(size_t)n * DUNITS + (k - DUNITS)];
    Wc[i] = (__bf16)v;
}

__global__ void convert_bf16_k(const float* __restrict__ src, __bf16* __restrict__ dst, size_t n) {
    size_t i = (size_t)blockIdx.x * blockDim.x + threadIdx.x;
    if (i < n) dst[i] = (__bf16)src[i];
}

// ============================================================
// pre_enc[b,t,a] = tanh(hs_pad[b,t,:] . Wenc[a,:] + benc[a])   (one-time)
// ============================================================
__global__ void preenc_k(const float* __restrict__ hs, const float* __restrict__ Wenc,
                         const float* __restrict__ benc, float* __restrict__ pre_enc) {
    int row = blockIdx.x;              // b*T + t  (0..16383)
    int tid = threadIdx.x;             // 256
    __shared__ float hsh[EPROJS];
    const float* h = hs + (size_t)row * EPROJS;
    for (int d = tid; d < EPROJS; d += 256) hsh[d] = h[d];
    __syncthreads();
    for (int a = tid; a < ATT; a += 256) {
        const float* w = Wenc + (size_t)a * EPROJS;
        float s = benc[a];
        for (int d = 0; d < EPROJS; ++d) s += hsh[d] * w[d];
        pre_enc[(size_t)row * ATT + a] = tanhf(s);
    }
}

// ============================================================
// per-step attention, fused: dec = tanh(z0 Wdec^T); e; masked softmax;
// att_c = w.hs; X0 = bf16[ ey_t | att_c | z0 ] ;  one block per batch row
// ============================================================
__global__ void attention_step_k(const float* __restrict__ z0, const float* __restrict__ pre_enc,
                                 const float* __restrict__ hs, const int* __restrict__ hlens,
                                 const int* __restrict__ ys_pad, const float* __restrict__ Wdec,
                                 const float* __restrict__ embed_w, __bf16* __restrict__ X0,
                                 int t) {
    int b = blockIdx.x;
    int tid = threadIdx.x;             // 256
    __shared__ float dec[ATT];
    __shared__ float wbuf[TT];
    __shared__ float red[8];

    const float* z = z0 + (size_t)b * DUNITS;
    for (int a = tid; a < ATT; a += 256) {
        const float* wd = Wdec + (size_t)a * DUNITS;
        float s = 0.f;
        for (int d = 0; d < DUNITS; ++d) s += z[d] * wd[d];
        dec[a] = tanhf(s);
    }
    __syncthreads();

    int hl = hlens[b];
    for (int tt = tid; tt < TT; tt += 256) {
        const float* pe = pre_enc + ((size_t)b * TT + tt) * ATT;
        float s = 0.f;
        for (int a = 0; a < ATT; ++a) s += pe[a] * dec[a];
        wbuf[tt] = (tt < hl) ? 2.0f * s : -1e10f;
    }
    __syncthreads();

    // block softmax over T=512
    float lm = -3.0e38f;
    for (int tt = tid; tt < TT; tt += 256) lm = fmaxf(lm, wbuf[tt]);
    for (int off = 16; off > 0; off >>= 1) lm = fmaxf(lm, __shfl_xor(lm, off));
    if ((tid & 31) == 0) red[tid >> 5] = lm;
    __syncthreads();
    float bm = red[0];
    for (int i = 1; i < 8; ++i) bm = fmaxf(bm, red[i]);
    __syncthreads();

    float ls = 0.f;
    for (int tt = tid; tt < TT; tt += 256) { float e = __expf(wbuf[tt] - bm); wbuf[tt] = e; ls += e; }
    for (int off = 16; off > 0; off >>= 1) ls += __shfl_xor(ls, off);
    if ((tid & 31) == 0) red[tid >> 5] = ls;
    __syncthreads();
    float bs = 0.f;
    for (int i = 0; i < 8; ++i) bs += red[i];
    float inv = 1.0f / bs;
    __syncthreads();
    for (int tt = tid; tt < TT; tt += 256) wbuf[tt] *= inv;
    __syncthreads();

    // att_c -> X0[1024..1535]
    for (int d = tid; d < EPROJS; d += 256) {
        const float* h = hs + (size_t)b * TT * EPROJS + d;
        float s = 0.f;
        for (int tt = 0; tt < TT; ++tt) s += wbuf[tt] * h[(size_t)tt * EPROJS];
        X0[(size_t)b * K0CAT + DUNITS + d] = (__bf16)s;
    }
    // ey_t -> X0[0..1023] ; z0 carry -> X0[1536..2559]
    int tok = (t == 0) ? SOS_EOS : ys_pad[b * LL + (t - 1)];
    for (int d = tid; d < DUNITS; d += 256) {
        X0[(size_t)b * K0CAT + d] = (__bf16)embed_w[(size_t)tok * DUNITS + d];
        X0[(size_t)b * K0CAT + (DUNITS + EPROJS) + d] = (__bf16)z[d];
    }
}

// ============================================================
// WMMA GEMM: Gpart[ks][M,N] = X[M, ks-quarter of K] * W[N,K]^T  (bf16->f32)
// one wave owns a 16x64 strip (4 accumulators sharing the A fragment, 4-way
// wmma ILP) over one quarter of K; 4 partial planes summed in pointwise.
// ============================================================
__global__ void gemm_xwT_wmma4_k(const __bf16* __restrict__ X, const __bf16* __restrict__ W,
                                 float* __restrict__ Gpart, int M, int N, int K) {
    int wid  = (int)((blockIdx.x * blockDim.x + threadIdx.x) >> 5);
    int lane = threadIdx.x & 31;
    int ngn = N >> 6;                          // 64-wide n-strips
    int nwaves = (M >> 4) * ngn * 4;           // x4 K-split
    if (wid >= nwaves) return;
    int ks = wid & 3;
    int ng = (wid >> 2) % ngn;
    int mt = (wid >> 2) / ngn;
    int m0 = mt << 4, n0 = ng << 6;
    int half = lane >> 4, l16 = lane & 15;
    int kq = K >> 2;
    int kbeg = ks * kq;

    const __bf16* xrow = X + (size_t)(m0 + l16) * K + (half ? 16 : 0) + kbeg;
    const __bf16* wr0 = W + (size_t)(n0 +  0 + l16) * K + (half ? 16 : 0) + kbeg;
    const __bf16* wr1 = W + (size_t)(n0 + 16 + l16) * K + (half ? 16 : 0) + kbeg;
    const __bf16* wr2 = W + (size_t)(n0 + 32 + l16) * K + (half ? 16 : 0) + kbeg;
    const __bf16* wr3 = W + (size_t)(n0 + 48 + l16) * K + (half ? 16 : 0) + kbeg;

    v8f a0 = {}, a1 = {}, a2 = {}, a3 = {};
#pragma unroll 4
    for (int k0 = 0; k0 < kq; k0 += 32) {
        v16bf a  = load_frag(xrow + k0, xrow + k0 + 8);
        v16bf b0 = load_frag(wr0 + k0, wr0 + k0 + 8);
        v16bf b1 = load_frag(wr1 + k0, wr1 + k0 + 8);
        v16bf b2 = load_frag(wr2 + k0, wr2 + k0 + 8);
        v16bf b3 = load_frag(wr3 + k0, wr3 + k0 + 8);
        a0 = __builtin_amdgcn_wmma_f32_16x16x32_bf16(false, a, false, b0, (short)0, a0, false, false);
        a1 = __builtin_amdgcn_wmma_f32_16x16x32_bf16(false, a, false, b1, (short)0, a1, false, false);
        a2 = __builtin_amdgcn_wmma_f32_16x16x32_bf16(false, a, false, b2, (short)0, a2, false, false);
        a3 = __builtin_amdgcn_wmma_f32_16x16x32_bf16(false, a, false, b3, (short)0, a3, false, false);
    }
    float* Gp = Gpart + (size_t)ks * M * N;
    int rowb = m0 + (half ? 8 : 0);
#pragma unroll
    for (int r = 0; r < 8; ++r) {
        size_t base = (size_t)(rowb + r) * N + l16;
        Gp[base + n0]      = a0[r];
        Gp[base + n0 + 16] = a1[r];
        Gp[base + n0 + 32] = a2[r];
        Gp[base + n0 + 48] = a3[r];
    }
}

// ============================================================
// LSTM pointwise gate math (torch gate order i,f,g,o); sums 4 K-split planes
// ============================================================
__device__ __forceinline__ float gsum4(const float* gp, int o) {
    return gp[o] + gp[o + GPLANE] + gp[o + 2 * GPLANE] + gp[o + 3 * GPLANE];
}

__global__ void lstm0_pw_k(const float* __restrict__ g0, const float* __restrict__ bih,
                           const float* __restrict__ bhh, float* __restrict__ z0,
                           float* __restrict__ c0, const float* __restrict__ z1,
                           __bf16* __restrict__ X1) {
    int idx = blockIdx.x * blockDim.x + threadIdx.x;
    if (idx >= BB * DUNITS) return;
    int b = idx >> 10, j = idx & (DUNITS - 1);
    const float* gp = g0 + (size_t)b * G4;
    float gi = gsum4(gp, j)              + bih[j]              + bhh[j];
    float gf = gsum4(gp, j + DUNITS)     + bih[j + DUNITS]     + bhh[j + DUNITS];
    float gg = gsum4(gp, j + 2 * DUNITS) + bih[j + 2 * DUNITS] + bhh[j + 2 * DUNITS];
    float go = gsum4(gp, j + 3 * DUNITS) + bih[j + 3 * DUNITS] + bhh[j + 3 * DUNITS];
    float c2 = sigmoidf_(gf) * c0[idx] + sigmoidf_(gi) * tanhf(gg);
    float h2 = sigmoidf_(go) * tanhf(c2);
    c0[idx] = c2;
    z0[idx] = h2;
    X1[(size_t)b * K1CAT + j]          = (__bf16)h2;       // new z0
    X1[(size_t)b * K1CAT + DUNITS + j] = (__bf16)z1[idx];  // z1 carry
}

__global__ void lstm1_pw_k(const float* __restrict__ g1, const float* __restrict__ bih,
                           const float* __restrict__ bhh, float* __restrict__ z1,
                           float* __restrict__ c1, __bf16* __restrict__ zs, int t) {
    int idx = blockIdx.x * blockDim.x + threadIdx.x;
    if (idx >= BB * DUNITS) return;
    int b = idx >> 10, j = idx & (DUNITS - 1);
    const float* gp = g1 + (size_t)b * G4;
    float gi = gsum4(gp, j)              + bih[j]              + bhh[j];
    float gf = gsum4(gp, j + DUNITS)     + bih[j + DUNITS]     + bhh[j + DUNITS];
    float gg = gsum4(gp, j + 2 * DUNITS) + bih[j + 2 * DUNITS] + bhh[j + 2 * DUNITS];
    float go = gsum4(gp, j + 3 * DUNITS) + bih[j + 3 * DUNITS] + bhh[j + 3 * DUNITS];
    float c2 = sigmoidf_(gf) * c1[idx] + sigmoidf_(gi) * tanhf(gg);
    float h2 = sigmoidf_(go) * tanhf(c2);
    c1[idx] = c2;
    z1[idx] = h2;
    zs[((size_t)t * BB + b) * DUNITS + j] = (__bf16)h2;
}

// ============================================================
// output projection + online log-softmax + NLL + argmax (no logits tensor)
// block = 8 waves handles one 16-row M-tile; each wave strides groups of
// 4 N-tiles (shared A fragment, 4 wmma accumulators)
// ============================================================
__global__ void logits_loss_k(const __bf16* __restrict__ zs, const __bf16* __restrict__ Wout,
                              const float* __restrict__ bout, const int* __restrict__ ys_pad,
                              float* __restrict__ accum) {
    const int K = DUNITS, NT = ODIM / 16;       // 625 n-tiles
    const int NGRP = (NT + 3) / 4;              // 157 groups of 4
    int tid = threadIdx.x;
    int wave = tid >> 5, lane = tid & 31;
    int half = lane >> 4, l16 = lane & 15;
    int m0 = blockIdx.x << 4;

    __shared__ int   tgt[16];
    __shared__ float sm_m[8][16], sm_s[8][16], sm_t[8][16], sm_av[8][16];
    __shared__ int   sm_ac[8][16];

    if (tid < 16) {
        int row = m0 + tid;               // row = t*32 + b
        int t = row >> 5, b = row & 31;
        tgt[tid] = (t < LL) ? ys_pad[b * LL + t] : SOS_EOS;   // ys_out
    }
    __syncthreads();

    float m_run[8], s_run[8], t_run[8], av_run[8];
    int ac_run[8];
#pragma unroll
    for (int r = 0; r < 8; ++r) {
        m_run[r] = -3.0e38f; s_run[r] = 0.f; t_run[r] = -3.0e38f;
        av_run[r] = -3.0e38f; ac_run[r] = 0;
    }

    const __bf16* xrow = zs + (size_t)(m0 + l16) * K + (half ? 16 : 0);
    for (int g = wave; g < NGRP; g += 8) {
        int base_nt = g << 2;
        const __bf16* wr[4];
#pragma unroll
        for (int j = 0; j < 4; ++j) {
            int nt = base_nt + j; if (nt >= NT) nt = NT - 1;   // clamp (stats masked)
            wr[j] = Wout + ((size_t)(nt << 4) + l16) * K + (half ? 16 : 0);
        }
        v8f acc[4];
        v8f zero = {};
#pragma unroll
        for (int j = 0; j < 4; ++j) acc[j] = zero;

#pragma unroll 2
        for (int k0 = 0; k0 < K; k0 += 32) {
            v16bf a  = load_frag(xrow + k0, xrow + k0 + 8);
            v16bf b0 = load_frag(wr[0] + k0, wr[0] + k0 + 8);
            v16bf b1 = load_frag(wr[1] + k0, wr[1] + k0 + 8);
            v16bf b2 = load_frag(wr[2] + k0, wr[2] + k0 + 8);
            v16bf b3 = load_frag(wr[3] + k0, wr[3] + k0 + 8);
            acc[0] = __builtin_amdgcn_wmma_f32_16x16x32_bf16(false, a, false, b0, (short)0, acc[0], false, false);
            acc[1] = __builtin_amdgcn_wmma_f32_16x16x32_bf16(false, a, false, b1, (short)0, acc[1], false, false);
            acc[2] = __builtin_amdgcn_wmma_f32_16x16x32_bf16(false, a, false, b2, (short)0, acc[2], false, false);
            acc[3] = __builtin_amdgcn_wmma_f32_16x16x32_bf16(false, a, false, b3, (short)0, acc[3], false, false);
        }

#pragma unroll
        for (int j = 0; j < 4; ++j) {
            int nt = base_nt + j;
            if (nt >= NT) break;                  // uniform across wave
            int col = (nt << 4) + l16;
            float bias = bout[col];
#pragma unroll
            for (int r = 0; r < 8; ++r) {
                float val = acc[j][r] + bias;
                int trow = tgt[r + half * 8];
                float cmax = val; int cidx = col;
                float tval = (col == trow) ? val : -3.0e38f;
                // reduce across the 16 columns of this half-wave (xor masks < 16)
                for (int off = 1; off < 16; off <<= 1) {
                    float om = __shfl_xor(cmax, off);
                    int   oi = __shfl_xor(cidx, off);
                    if (om > cmax || (om == cmax && oi < cidx)) { cmax = om; cidx = oi; }
                    tval = fmaxf(tval, __shfl_xor(tval, off));
                }
                float e = __expf(val - cmax);
                float csum = e;
                for (int off = 1; off < 16; off <<= 1) csum += __shfl_xor(csum, off);
                // online merge into running stats
                float nm = fmaxf(m_run[r], cmax);
                s_run[r] = s_run[r] * __expf(m_run[r] - nm) + csum * __expf(cmax - nm);
                m_run[r] = nm;
                t_run[r] = fmaxf(t_run[r], tval);
                if (cmax > av_run[r] || (cmax == av_run[r] && cidx < ac_run[r])) {
                    av_run[r] = cmax; ac_run[r] = cidx;
                }
            }
        }
    }
    if (l16 == 0) {
#pragma unroll
        for (int r = 0; r < 8; ++r) {
            int rl = r + half * 8;
            sm_m[wave][rl] = m_run[r]; sm_s[wave][rl] = s_run[r];
            sm_t[wave][rl] = t_run[r]; sm_av[wave][rl] = av_run[r]; sm_ac[wave][rl] = ac_run[r];
        }
    }
    __syncthreads();
    if (tid < 16) {
        float m = -3.0e38f, s = 0.f, tv = -3.0e38f, av = -3.0e38f; int ai = 0;
        for (int w = 0; w < 8; ++w) {
            float wm = sm_m[w][tid], ws = sm_s[w][tid];
            float nm = fmaxf(m, wm);
            s = s * __expf(m - nm) + ws * __expf(wm - nm);
            m = nm;
            tv = fmaxf(tv, sm_t[w][tid]);
            float wav = sm_av[w][tid]; int wai = sm_ac[w][tid];
            if (wav > av || (wav == av && wai < ai)) { av = wav; ai = wai; }
        }
        float logZ = m + __logf(s);
        float nll = logZ - tv;
        atomicAdd(&accum[0], nll);
        atomicAdd(&accum[1], (ai == tgt[tid]) ? 1.0f : 0.0f);
    }
}

__global__ void finalize_k(const float* __restrict__ accum, float* __restrict__ out) {
    float loss = accum[0] / (float)NROWS * (float)LL;
    out[0] = loss;
    out[1] = accum[1] / (float)NROWS;
    out[2] = __expf(loss / (float)BB);
}

// ============================================================
// host launcher
// ============================================================
static inline size_t align_up(size_t x) { return (x + 255) & ~(size_t)255; }

extern "C" void kernel_launch(void* const* d_in, const int* in_sizes, int n_in,
                              void* d_out, int out_size, void* d_ws, size_t ws_size,
                              hipStream_t stream) {
    // inputs per setup_inputs order
    const float* hs_pad  = (const float*)d_in[0];
    const int*   hlens   = (const int*)  d_in[1];
    const int*   ys_pad  = (const int*)  d_in[2];
    const float* embed_w = (const float*)d_in[3];
    const float* Wenc    = (const float*)d_in[4];
    const float* benc    = (const float*)d_in[5];
    const float* Wdec    = (const float*)d_in[6];
    const float* W_ih0   = (const float*)d_in[7];
    const float* W_hh0   = (const float*)d_in[8];
    const float* b_ih0   = (const float*)d_in[9];
    const float* b_hh0   = (const float*)d_in[10];
    const float* W_ih1   = (const float*)d_in[11];
    const float* W_hh1   = (const float*)d_in[12];
    const float* b_ih1   = (const float*)d_in[13];
    const float* b_hh1   = (const float*)d_in[14];
    const float* Wout    = (const float*)d_in[15];
    const float* bout    = (const float*)d_in[16];
    float* out = (float*)d_out;

    // workspace carve-out
    char* ws = (char*)d_ws;
    size_t off = 0;
    auto take = [&](size_t bytes) { char* p = ws + off; off += align_up(bytes); return p; };

    __bf16* W0cat   = (__bf16*)take((size_t)G4 * K0CAT * 2);        // 21.0 MB
    __bf16* W1cat   = (__bf16*)take((size_t)G4 * K1CAT * 2);        // 16.8 MB
    __bf16* WoutBf  = (__bf16*)take((size_t)ODIM * DUNITS * 2);     // 20.5 MB
    float*  pre_enc = (float*) take((size_t)BB * TT * ATT * 4);     // 21.0 MB
    __bf16* zsBf    = (__bf16*)take((size_t)NROWS * DUNITS * 2);    //  8.5 MB
    float*  z0      = (float*) take((size_t)BB * DUNITS * 4);
    float*  c0      = (float*) take((size_t)BB * DUNITS * 4);
    float*  z1      = (float*) take((size_t)BB * DUNITS * 4);
    float*  c1      = (float*) take((size_t)BB * DUNITS * 4);
    __bf16* X0      = (__bf16*)take((size_t)BB * K0CAT * 2);
    __bf16* X1      = (__bf16*)take((size_t)BB * K1CAT * 2);
    float*  g0      = (float*) take((size_t)4 * GPLANE * 4);        //  2 MB (4 K-split planes)
    float*  g1      = (float*) take((size_t)4 * GPLANE * 4);        //  2 MB
    float*  accum   = (float*) take(2 * 4);
    (void)ws_size; (void)in_sizes; (void)n_in; (void)out_size;

    // one-time (per launch) prep
    init_state_k<<<(BB * DUNITS + 255) / 256, 256, 0, stream>>>(z0, c0, z1, c1, accum);
    build_w0cat_k<<<(int)(((size_t)G4 * K0CAT + 255) / 256), 256, 0, stream>>>(W_ih0, W_hh0, W0cat);
    build_w1cat_k<<<(int)(((size_t)G4 * K1CAT + 255) / 256), 256, 0, stream>>>(W_ih1, W_hh1, W1cat);
    convert_bf16_k<<<(int)(((size_t)ODIM * DUNITS + 255) / 256), 256, 0, stream>>>(
        Wout, WoutBf, (size_t)ODIM * DUNITS);
    preenc_k<<<BB * TT, 256, 0, stream>>>(hs_pad, Wenc, benc, pre_enc);

    // sequential decode: 129 steps
    // waves = (M/16) * (N/64) * 4 K-splits = 2*64*4 = 512 -> 64 blocks of 8 waves
    const int gemm_waves  = (BB / 16) * (G4 / 64) * 4;
    const int gemm_blocks = (gemm_waves * 32 + 255) / 256;
    for (int t = 0; t < LP1; ++t) {
        attention_step_k<<<BB, 256, 0, stream>>>(z0, pre_enc, hs_pad, hlens, ys_pad,
                                                 Wdec, embed_w, X0, t);
        gemm_xwT_wmma4_k<<<gemm_blocks, 256, 0, stream>>>(X0, W0cat, g0, BB, G4, K0CAT);
        lstm0_pw_k<<<(BB * DUNITS + 255) / 256, 256, 0, stream>>>(g0, b_ih0, b_hh0, z0, c0, z1, X1);
        gemm_xwT_wmma4_k<<<gemm_blocks, 256, 0, stream>>>(X1, W1cat, g1, BB, G4, K1CAT);
        lstm1_pw_k<<<(BB * DUNITS + 255) / 256, 256, 0, stream>>>(g1, b_ih1, b_hh1, z1, c1, zsBf, t);
    }

    // output projection + fused online log-softmax / NLL / argmax
    logits_loss_k<<<NROWS / 16, 256, 0, stream>>>(zsBf, WoutBf, bout, ys_pad, accum);
    finalize_k<<<1, 1, 0, stream>>>(accum, out);
}
